// TGNetworkBias_25031069401274
// MI455X (gfx1250) — matrix operand order repaired and checked by reading, not compiled, not run on hardware
//
#include <hip/hip_runtime.h>

#define BN_EPS 1e-5f

typedef _Float16 v8h  __attribute__((ext_vector_type(8)));
typedef _Float16 v16h __attribute__((ext_vector_type(16)));
typedef float    v8f  __attribute__((ext_vector_type(8)));
typedef int      v4i  __attribute__((ext_vector_type(4)));

typedef __attribute__((address_space(1))) v4i* g_v4i_ptr;   // global (AS1) int4*
typedef __attribute__((address_space(3))) v4i* l_v4i_ptr;   // LDS (AS3) int4*

// ---------------------------------------------------------------------------
// CDNA5 async global->LDS copy (16B per lane), tracked by ASYNCcnt.
// Builtin signature (per hipcc diagnostic): param0 = int4 in global AS,
// param1 = int4 in LDS AS, then imm offset + imm cpol.
// ---------------------------------------------------------------------------
__device__ __forceinline__ void async_copy16(const _Float16* g, _Float16* l) {
#if __has_builtin(__builtin_amdgcn_global_load_async_to_lds_b128)
    __builtin_amdgcn_global_load_async_to_lds_b128(
        (g_v4i_ptr)(void*)g, (l_v4i_ptr)(void*)l, 0, 0);
#else
    unsigned loff = (unsigned)(size_t)(__attribute__((address_space(3))) _Float16*)l;
    asm volatile("global_load_async_to_lds_b128 %0, %1, off" :: "v"(loff), "v"(g) : "memory");
#endif
}

template<int N>
__device__ __forceinline__ void wait_asynccnt() {
#if __has_builtin(__builtin_amdgcn_s_wait_asynccnt)
    __builtin_amdgcn_s_wait_asynccnt(N);
#else
    asm volatile("s_wait_asynccnt %0" :: "i"(N));
#endif
}

__device__ __forceinline__ v16h combine16(v8h lo, v8h hi) {
    return __builtin_shufflevector(lo, hi, 0,1,2,3,4,5,6,7,8,9,10,11,12,13,14,15);
}

// ---------------------------------------------------------------------------
// f32 -> f16 conversion (feeds the WMMA path)
// ---------------------------------------------------------------------------
__global__ void cvt_f32_f16(const float* __restrict__ s, _Float16* __restrict__ d, int n) {
    int i = blockIdx.x * blockDim.x + threadIdx.x;
    if (i < n) d[i] = (_Float16)s[i];
}

// ---------------------------------------------------------------------------
// Epilogue helper.
// MODE 0: y = relu(acc*scale + bias) stored as f16   (BN folded per column)
// MODE 1: y = (clip(1.2*sigmoid(acc)-0.1,0,1) > 0.49999) ? 1 : 0, stored f32
// MODE 2: y = acc, stored f32
// ---------------------------------------------------------------------------
template<int MODE>
__device__ __forceinline__ void store_tile(v8f cc, int row0, int col, void* Cout, int ldc,
                                           const float* bng, const float* bnb,
                                           const float* bnm, const float* bnv) {
    if (MODE == 0) {
        float scale = bng[col] * rsqrtf(bnv[col] + BN_EPS);
        float bias  = bnb[col] - bnm[col] * scale;
        _Float16* C = (_Float16*)Cout;
#pragma unroll
        for (int i = 0; i < 8; ++i) {
            float v = fmaxf(cc[i] * scale + bias, 0.0f);
            C[(size_t)(row0 + i) * ldc + col] = (_Float16)v;
        }
    } else {
        float* C = (float*)Cout;
#pragma unroll
        for (int i = 0; i < 8; ++i) {
            float v = cc[i];
            if (MODE == 1) {
                float s = 1.0f / (1.0f + __expf(-v));
                s = fminf(fmaxf(1.2f * s - 0.1f, 0.0f), 1.0f);
                v = (s > 0.49999f) ? 1.0f : 0.0f;
            }
            C[(size_t)(row0 + i) * ldc + col] = v;
        }
    }
}

// ---------------------------------------------------------------------------
// Fused GEMM with async-LDS double buffering + v_wmma_f32_16x16x32_f16.
//   C[m][n] = sum_k A[m][k] * Bw[n][k]      (Bw row-major N x K, i.e. B = Bw^T)
// WG = 256 threads = 8 waves; WG tile 128(M) x 64(N); K-step 64 (two WMMA
// sub-steps of K=32); LDS double buffer, 6 async b128 copies per stage.
// Wave tile 32x32: 2 A frags x 2 B frags -> 4 WMMA per sub-step.
// All dims used here are multiples of the tile sizes -> no bounds checks.
// ---------------------------------------------------------------------------
template<int MODE>
__global__ __launch_bounds__(256) void gemm_bn_wmma_async(
    const _Float16* __restrict__ A,  int lda,
    const _Float16* __restrict__ Bw, int ldb,
    void* __restrict__ Cout, int ldc, int Kdim,
    const float* __restrict__ bng, const float* __restrict__ bnb,
    const float* __restrict__ bnm, const float* __restrict__ bnv)
{
    constexpr int LSTR  = 72;                 // halfs per LDS row (64 + 8 pad = 36 dwords)
    constexpr int AROWS = 128, BROWS = 64;
    constexpr int BBASE = AROWS * LSTR;
    constexpr int BUFH  = (AROWS + BROWS) * LSTR;   // 13824 halfs = 27648 B per buffer
    __shared__ _Float16 smem[2][BUFH];              // ~54 KB of the 320 KB WGP LDS

    const int tid  = threadIdx.x;
    const int lane = tid & 31;
    const int w    = tid >> 5;
    const int wm   = w & 3;                   // 4 M-strips of 32
    const int wn   = w >> 2;                  // 2 N-strips of 32
    const int r    = lane & 15;
    const int half = lane >> 4;
    const int rowBase = blockIdx.y * 128;
    const int colBase = blockIdx.x * 64;

    // copy indexing: each lane moves one 16B chunk per instruction
    const int crow = tid >> 3;                // 0..31
    const int cchk = tid & 7;                 // 0..7  (8 chunks x 8 halfs = 64 cols)

    const _Float16* gA = A  + (size_t)(rowBase + crow) * lda + cchk * 8;
    const _Float16* gB = Bw + (size_t)(colBase + crow) * ldb + cchk * 8;

    auto issue = [&](int buf, int k0) {
        _Float16* la = &smem[buf][crow * LSTR + cchk * 8];
        _Float16* lb = &smem[buf][BBASE + crow * LSTR + cchk * 8];
#pragma unroll
        for (int i = 0; i < 4; ++i)           // A tile: 128 rows
            async_copy16(gA + (size_t)(32 * i) * lda + k0, la + 32 * i * LSTR);
#pragma unroll
        for (int i = 0; i < 2; ++i)           // B tile: 64 rows
            async_copy16(gB + (size_t)(32 * i) * ldb + k0, lb + 32 * i * LSTR);
    };

    v8f c00 = {}, c01 = {}, c10 = {}, c11 = {};

    issue(0, 0);
    int cur = 0;
    for (int k0 = 0; k0 < Kdim; k0 += 64) {
        if (k0 + 64 < Kdim) {
            issue(cur ^ 1, k0 + 64);          // prefetch next stage (async)
            wait_asynccnt<6>();               // current stage's 6 copies complete
        } else {
            wait_asynccnt<0>();
        }
        __syncthreads();                      // publish LDS across all 8 waves

        const _Float16* As = &smem[cur][0];
        const _Float16* Bs = &smem[cur][BBASE];
#pragma unroll
        for (int ks = 0; ks < 64; ks += 32) {
            // A frag (16x32): lane holds row, chunks at ks+half*8 and ks+16+half*8
            const _Float16* pa0 = As + (wm * 32 + r) * LSTR + ks + half * 8;
            const _Float16* pa1 = pa0 + 16 * LSTR;
            v16h a0 = combine16(*(const v8h*)pa0, *(const v8h*)(pa0 + 16));
            v16h a1 = combine16(*(const v8h*)pa1, *(const v8h*)(pa1 + 16));
            // B frag (32x16): lane holds col row of Bw, chunks at ks+half*8 and ks+16+half*8
            const _Float16* pb0 = Bs + (wn * 32 + r) * LSTR + ks + half * 8;
            const _Float16* pb1 = pb0 + 16 * LSTR;
            v16h b0 = combine16(*(const v8h*)pb0, *(const v8h*)(pb0 + 16));
            v16h b1 = combine16(*(const v8h*)pb1, *(const v8h*)(pb1 + 16));

            c00 = __builtin_amdgcn_wmma_f32_16x16x32_f16(false, a0, false, b0, (short)0, c00, false, false);
            c01 = __builtin_amdgcn_wmma_f32_16x16x32_f16(false, a0, false, b1, (short)0, c01, false, false);
            c10 = __builtin_amdgcn_wmma_f32_16x16x32_f16(false, a1, false, b0, (short)0, c10, false, false);
            c11 = __builtin_amdgcn_wmma_f32_16x16x32_f16(false, a1, false, b1, (short)0, c11, false, false);
        }
        __syncthreads();                      // all reads done before buffer reuse
        cur ^= 1;
    }

    // C layout: element i -> row (i + 8*half), col (lane&15)
    const int rb = rowBase + wm * 32, cb = colBase + wn * 32;
    store_tile<MODE>(c00, rb +  0 + half * 8, cb +  0 + r, Cout, ldc, bng, bnb, bnm, bnv);
    store_tile<MODE>(c01, rb +  0 + half * 8, cb + 16 + r, Cout, ldc, bng, bnb, bnm, bnv);
    store_tile<MODE>(c10, rb + 16 + half * 8, cb +  0 + r, Cout, ldc, bng, bnb, bnm, bnv);
    store_tile<MODE>(c11, rb + 16 + half * 8, cb + 16 + r, Cout, ldc, bng, bnb, bnm, bnv);
}

// ---------------------------------------------------------------------------
// Host-side orchestration
// ---------------------------------------------------------------------------
extern "C" void kernel_launch(void* const* d_in, const int* in_sizes, int n_in,
                              void* d_out, int out_size, void* d_ws, size_t ws_size,
                              hipStream_t stream)
{
    const int B = 2048, D = 1024, G = 16, S = 128, Kk = 128, E = 8, H = 512;
    const int d1 = G * 8, d2 = G * 64, d3 = G * 128, d4 = G * S;  // 128,1024,2048,2048

    const float* prompt = (const float*)d_in[0];
    const float* tw1 = (const float*)d_in[1];
    const float* tg1 = (const float*)d_in[2],  *tb1 = (const float*)d_in[3];
    const float* tm1 = (const float*)d_in[4],  *tv1 = (const float*)d_in[5];
    const float* tw2 = (const float*)d_in[6];
    const float* tg2 = (const float*)d_in[7],  *tb2 = (const float*)d_in[8];
    const float* tm2 = (const float*)d_in[9],  *tv2 = (const float*)d_in[10];
    const float* tw3 = (const float*)d_in[11];
    const float* tg3 = (const float*)d_in[12], *tb3 = (const float*)d_in[13];
    const float* tm3 = (const float*)d_in[14], *tv3 = (const float*)d_in[15];
    const float* tw4 = (const float*)d_in[16];
    const float* tg4 = (const float*)d_in[17], *tb4 = (const float*)d_in[18];
    const float* tm4 = (const float*)d_in[19], *tv4 = (const float*)d_in[20];
    const float* ew1 = (const float*)d_in[21];
    const float* ew2 = (const float*)d_in[22];
    const float* ebg = (const float*)d_in[23], *ebb = (const float*)d_in[24];
    const float* ebm = (const float*)d_in[25], *ebv = (const float*)d_in[26];

    // Workspace sub-allocations (256B aligned); total ~41 MB
    char* wsp = (char*)d_ws;
    size_t off = 0;
    auto walloc = [&](size_t bytes) -> void* {
        void* p = wsp + off;
        off += (bytes + 255) & ~(size_t)255;
        return p;
    };
    _Float16* hp  = (_Float16*)walloc((size_t)B  * D  * 2);
    _Float16* hw1 = (_Float16*)walloc((size_t)d1 * D  * 2);
    _Float16* hw2 = (_Float16*)walloc((size_t)d2 * d1 * 2);
    _Float16* hw3 = (_Float16*)walloc((size_t)d3 * d2 * 2);
    _Float16* hw4 = (_Float16*)walloc((size_t)d4 * d3 * 2);
    _Float16* he1 = (_Float16*)walloc((size_t)E * H * S * 2);
    _Float16* he2 = (_Float16*)walloc((size_t)E * Kk * H * 2);
    _Float16* x1  = (_Float16*)walloc((size_t)B * d1 * 2);
    _Float16* x2  = (_Float16*)walloc((size_t)B * d2 * 2);
    _Float16* x3  = (_Float16*)walloc((size_t)B * d3 * 2);
    _Float16* x4  = (_Float16*)walloc((size_t)B * d4 * 2);
    _Float16* hb  = (_Float16*)walloc((size_t)B * H * 2);   // reused per (e,g)

    auto cvt = [&](const float* s, _Float16* d, int n) {
        cvt_f32_f16<<<dim3((n + 255) / 256), dim3(256), 0, stream>>>(s, d, n);
    };
    cvt(prompt, hp,  B * D);
    cvt(tw1,    hw1, d1 * D);
    cvt(tw2,    hw2, d2 * d1);
    cvt(tw3,    hw3, d3 * d2);
    cvt(tw4,    hw4, d4 * d3);
    cvt(ew1,    he1, E * H * S);
    cvt(ew2,    he2, E * Kk * H);

    // ---- trunk: 4x fused GEMM+BN+ReLU ----
    gemm_bn_wmma_async<0><<<dim3(d1/64, B/128), dim3(256), 0, stream>>>(hp, D,  hw1, D,  x1, d1, D,  tg1, tb1, tm1, tv1);
    gemm_bn_wmma_async<0><<<dim3(d2/64, B/128), dim3(256), 0, stream>>>(x1, d1, hw2, d1, x2, d2, d1, tg2, tb2, tm2, tv2);
    gemm_bn_wmma_async<0><<<dim3(d3/64, B/128), dim3(256), 0, stream>>>(x2, d2, hw3, d2, x3, d3, d2, tg3, tb3, tm3, tv3);
    gemm_bn_wmma_async<0><<<dim3(d4/64, B/128), dim3(256), 0, stream>>>(x3, d3, hw4, d3, x4, d4, d3, tg4, tb4, tm4, tv4);

    // ---- expert stage: only experts 0 (selection) and E-1 (bias) are used ----
    float* selp  = (float*)d_out;                    // (B, G, K) flat
    float* biasp = selp + (size_t)B * G * Kk;        // (B, G, K) flat

    for (int g = 0; g < G; ++g) {
        const size_t bnoff0 = ((size_t)0 * G + g) * H;
        const size_t bnoff7 = ((size_t)(E - 1) * G + g) * H;

        // expert 0: h = relu(bn(le[:,g,:] @ ew1[0]^T)) ; sel = thresh(sigmoid(h @ ew2[0]^T))
        gemm_bn_wmma_async<0><<<dim3(H/64, B/128), dim3(256), 0, stream>>>(
            x4 + g * S, d4, he1 + (size_t)0 * H * S, S, hb, H, S,
            ebg + bnoff0, ebb + bnoff0, ebm + bnoff0, ebv + bnoff0);
        gemm_bn_wmma_async<1><<<dim3(Kk/64, B/128), dim3(256), 0, stream>>>(
            hb, H, he2 + (size_t)0 * Kk * H, H, selp + g * Kk, G * Kk, H,
            nullptr, nullptr, nullptr, nullptr);

        // expert E-1: layer_bias = h @ ew2[7]^T
        gemm_bn_wmma_async<0><<<dim3(H/64, B/128), dim3(256), 0, stream>>>(
            x4 + g * S, d4, he1 + (size_t)(E - 1) * H * S, S, hb, H, S,
            ebg + bnoff7, ebb + bnoff7, ebm + bnoff7, ebv + bnoff7);
        gemm_bn_wmma_async<2><<<dim3(Kk/64, B/128), dim3(256), 0, stream>>>(
            hb, H, he2 + (size_t)(E - 1) * Kk * H, H, biasp + g * Kk, G * Kk, H,
            nullptr, nullptr, nullptr, nullptr);
    }
}
